// StateEncoder_82592221102837
// MI455X (gfx1250) — matrix operand order repaired and checked by reading, not compiled, not run on hardware
//
#include <hip/hip_runtime.h>

typedef __attribute__((ext_vector_type(16))) __bf16 v16bf;
typedef __attribute__((ext_vector_type(8)))  __bf16 v8bf;
typedef __attribute__((ext_vector_type(8)))  float  v8f;

#define NEG_INF_ENC 0x007FFFFFu

__device__ __forceinline__ float leakyf(float x){ return x >= 0.f ? x : 0.2f * x; }

__device__ __forceinline__ unsigned short f2bf(float f){
  unsigned u = __float_as_uint(f);
  unsigned r = u + 0x7FFFu + ((u >> 16) & 1u);   // round-to-nearest-even
  return (unsigned short)(r >> 16);
}
// order-preserving float<->uint encoding for atomicMax on floats
__device__ __forceinline__ unsigned fenc(float f){
  unsigned u = __float_as_uint(f);
  return (u & 0x80000000u) ? ~u : (u | 0x80000000u);
}
__device__ __forceinline__ float fdec(unsigned e){
  unsigned u = (e & 0x80000000u) ? (e & 0x7FFFFFFFu) : ~e;
  return __uint_as_float(u);
}

// ---------------- generic fills / converts ----------------
__global__ void k_fill_f32(float* p, float v, long n){
  long i = (long)blockIdx.x * blockDim.x + threadIdx.x;
  if (i < n) p[i] = v;
}
__global__ void k_fill_u32(unsigned* p, unsigned v, long n){
  long i = (long)blockIdx.x * blockDim.x + threadIdx.x;
  if (i < n) p[i] = v;
}
__global__ void k_f32_to_bf16(const float* s, unsigned short* d, long n){
  long i = (long)blockIdx.x * blockDim.x + threadIdx.x;
  if (i < n) d[i] = f2bf(s[i]);
}
// Wt[n, k] = bf16(W[k, n]);  W is [K, Nc] row-major
__global__ void k_w_transpose_bf16(const float* W, unsigned short* Wt, int K, int Nc){
  long i = (long)blockIdx.x * blockDim.x + threadIdx.x;
  if (i >= (long)K * Nc) return;
  int n = (int)(i / K), k = (int)(i % K);
  Wt[i] = f2bf(W[(size_t)k * Nc + n]);
}

// ---------------- node feature build: x = [ent+p | rel | dep] ----------------
__global__ void k_build_x(const int* eidx, const int* ridx, const int* didx,
                          const float* avg, const float* embE, const float* embR,
                          const float* embD, const float* pw1, const float* pb1,
                          const float* pw2, const float* pb2, float* X, int N){
  long i = (long)blockIdx.x * blockDim.x + threadIdx.x;
  if (i >= (long)N * 288) return;
  int n = (int)(i / 288), f = (int)(i % 288);
  float v;
  if (f < 128) {
    float av = avg[n];
    float p = pb2[f];
    for (int k = 0; k < 64; ++k) {
      float t = leakyf(av * pw1[k] + pb1[k]);
      p += t * pw2[k * 128 + f];
    }
    v = embE[(size_t)eidx[n] * 128 + f] + p;
  } else if (f < 256) {
    v = embR[(size_t)ridx[n] * 128 + (f - 128)];
  } else {
    int d = didx[n]; if (d > 9) d = 9;
    v = embD[d * 32 + (f - 256)];
  }
  X[(size_t)n * 288 + f] = v;
}

// A-fragment loader per ISA 16-bit A layout:
// elems 0..7 -> K = k0 + 8g + e ; elems 8..15 -> K = k0 + 16 + 8g + (e-8)
__device__ __forceinline__ v16bf load_afrag(const __bf16* arow, int k0, int g){
  v8bf lo = *(const v8bf*)(arow + k0 + 8 * g);
  v8bf hi = *(const v8bf*)(arow + k0 + 16 + 8 * g);
  v16bf a;
#pragma unroll
  for (int q = 0; q < 8; ++q) { a[q] = lo[q]; a[q + 8] = hi[q]; }
  return a;
}

// ---------------- WMMA bf16 GEMM: C[M,Nc] = A[M,K] * Bt[Nc,K]^T ----------------
// One wave per 32x32 output tile (2x2 register blocking, 4 accumulators):
// per K-step loads 2 A-frags + 2 B-frags and issues 4 WMMAs (2x intensity vs 16x16).
// K % 32 == 0, Nc % 32 == 0. M tail (M%32==16) handled by clamped loads + guarded stores,
// keeping the WMMA region branch-free (EXEC all-ones).
__global__ void k_wmma_gemm(const unsigned short* Abf, const unsigned short* Btbf,
                            float* C, int M, int K, int Nc){
  int wave = (int)((blockIdx.x * (long)blockDim.x + threadIdx.x) >> 5);
  int lane = threadIdx.x & 31;
  int tilesN = Nc >> 5;
  int tilesM = (M + 31) >> 5;
  if (wave >= tilesM * tilesN) return;           // uniform within wave at runtime
  int tm = wave / tilesN, tn = wave % tilesN;
  int g = lane >> 4, r = lane & 15;
  int mBase = tm * 32, nBase = tn * 32;
  int mrow0 = mBase + r;
  int mrow1 = mBase + 16 + r; if (mrow1 > M - 1) mrow1 = M - 1;   // clamp tail reads
  const __bf16* A  = (const __bf16*)Abf;
  const __bf16* Bt = (const __bf16*)Btbf;
  const __bf16* arow0 = A  + (size_t)mrow0 * K;
  const __bf16* arow1 = A  + (size_t)mrow1 * K;
  const __bf16* brow0 = Bt + (size_t)(nBase + r) * K;
  const __bf16* brow1 = Bt + (size_t)(nBase + 16 + r) * K;
  v8f acc00 = {}, acc01 = {}, acc10 = {}, acc11 = {};
  for (int k0 = 0; k0 < K; k0 += 32) {
    v16bf a0 = load_afrag(arow0, k0, g);
    v16bf a1 = load_afrag(arow1, k0, g);
    // B fragment: same K layout as A (elems 0..7 -> k0 + 8g + e, 8..15 -> k0 + 16 + 8g + (e-8))
    v16bf b0 = load_afrag(brow0, k0, g);
    v16bf b1 = load_afrag(brow1, k0, g);
    acc00 = __builtin_amdgcn_wmma_f32_16x16x32_bf16(false, a0, false, b0, (short)0, acc00, false, false);
    acc01 = __builtin_amdgcn_wmma_f32_16x16x32_bf16(false, a0, false, b1, (short)0, acc01, false, false);
    acc10 = __builtin_amdgcn_wmma_f32_16x16x32_bf16(false, a1, false, b0, (short)0, acc10, false, false);
    acc11 = __builtin_amdgcn_wmma_f32_16x16x32_bf16(false, a1, false, b1, (short)0, acc11, false, false);
  }
#pragma unroll
  for (int v = 0; v < 8; ++v) {
    int m0 = mBase + v + 8 * g;        // rows of upper 16x32 strip
    int m1 = m0 + 16;                  // rows of lower 16x32 strip
    int n0 = nBase + r, n1 = nBase + 16 + r;
    if (m0 < M) {
      C[(size_t)m0 * Nc + n0] = acc00[v];
      C[(size_t)m0 * Nc + n1] = acc01[v];
    }
    if (m1 < M) {
      C[(size_t)m1 * Nc + n0] = acc10[v];
      C[(size_t)m1 * Nc + n1] = acc11[v];
    }
  }
}

// ---------------- attention ----------------
__global__ void k_scores(const float* Hf, const float* avec, float* out, int N, int H, int C){
  long i = (long)blockIdx.x * blockDim.x + threadIdx.x;
  if (i >= (long)N * H) return;
  int n = (int)(i / H), h = (int)(i % H);
  const float* row = Hf + (size_t)n * (H * C) + (size_t)h * C;
  const float* a = avec + (size_t)h * C;
  float s = 0.f;
  for (int c = 0; c < C; ++c) s += row[c] * a[c];
  out[i] = s;
}
__device__ __forceinline__ void edge_sd(const int* esrc, const int* edst, int E, int e,
                                        int& s, int& d){
  if (e < E) { s = esrc[e]; d = edst[e]; } else { s = e - E; d = e - E; }
}
__global__ void k_edge_max(const int* esrc, const int* edst, int E, int N, int H,
                           const float* as, const float* ds, unsigned* menc){
  long i = (long)blockIdx.x * blockDim.x + threadIdx.x;
  long tot = (long)(E + N) * H;
  if (i >= tot) return;
  int e = (int)(i / H), h = (int)(i % H), s, d;
  edge_sd(esrc, edst, E, e, s, d);
  float v = leakyf(as[(size_t)s * H + h] + ds[(size_t)d * H + h]);
  atomicMax(&menc[(size_t)d * H + h], fenc(v));
}
__global__ void k_edge_sum(const int* esrc, const int* edst, int E, int N, int H,
                           const float* as, const float* ds, const unsigned* menc, float* ssum){
  long i = (long)blockIdx.x * blockDim.x + threadIdx.x;
  long tot = (long)(E + N) * H;
  if (i >= tot) return;
  int e = (int)(i / H), h = (int)(i % H), s, d;
  edge_sd(esrc, edst, E, e, s, d);
  float v = leakyf(as[(size_t)s * H + h] + ds[(size_t)d * H + h]);
  float m = fdec(menc[(size_t)d * H + h]);
  if (!__builtin_isfinite(m)) m = 0.f;
  atomicAdd(&ssum[(size_t)d * H + h], __expf(v - m));
}
__global__ void k_edge_agg(const int* esrc, const int* edst, int E, int N, int H, int C,
                           const float* as, const float* ds, const unsigned* menc,
                           const float* ssum, const float* Hf, float* Out){
  long i = (long)blockIdx.x * blockDim.x + threadIdx.x;
  long tot = (long)(E + N) * H;
  if (i >= tot) return;
  int e = (int)(i / H), h = (int)(i % H), s, d;
  edge_sd(esrc, edst, E, e, s, d);
  float v = leakyf(as[(size_t)s * H + h] + ds[(size_t)d * H + h]);
  float m = fdec(menc[(size_t)d * H + h]);
  if (!__builtin_isfinite(m)) m = 0.f;
  float alpha = __expf(v - m) / (ssum[(size_t)d * H + h] + 1e-16f);
  const float* hs = Hf + (size_t)s * (H * C) + (size_t)h * C;
  float* od = Out + (size_t)d * (H * C) + (size_t)h * C;
  for (int c = 0; c < C; ++c) atomicAdd(&od[c], alpha * hs[c]);
}

// ---------------- layernorm (+bias, opt leaky, opt residual); one wave per node ----------
__global__ void k_ln(const float* in, const float* gbias, const float* g, const float* b,
                     const float* resid, float* out, int N, int F, int doLeaky, int doResid){
  int wave = (int)((blockIdx.x * (long)blockDim.x + threadIdx.x) >> 5);
  int lane = threadIdx.x & 31;
  if (wave >= N) return;
  const float* row = in + (size_t)wave * F;
  float sum = 0.f, sq = 0.f;
  for (int f = lane; f < F; f += 32) { float v = row[f] + gbias[f]; sum += v; sq += v * v; }
  for (int o = 16; o > 0; o >>= 1) { sum += __shfl_xor(sum, o, 32); sq += __shfl_xor(sq, o, 32); }
  float mu = sum / F;
  float var = sq / F - mu * mu;
  float inv = rsqrtf(var + 1e-5f);
  for (int f = lane; f < F; f += 32) {
    float v = (row[f] + gbias[f] - mu) * inv * g[f] + b[f];
    if (doLeaky) v = leakyf(v);
    if (doResid) v += resid[(size_t)wave * F + f];
    out[(size_t)wave * F + f] = v;
  }
}

// ---------------- pooling + projection ----------------
__global__ void k_pool_acc(const float* X2, const int* batch, float* cnt, float* macc,
                           unsigned* pmax, int N){
  long i = (long)blockIdx.x * blockDim.x + threadIdx.x;
  if (i >= (long)N * 128) return;
  int n = (int)(i / 128), f = (int)(i % 128);
  int bg = batch[n];
  float v = X2[(size_t)n * 128 + f];
  atomicAdd(&macc[(size_t)bg * 128 + f], v);
  atomicMax(&pmax[(size_t)bg * 128 + f], fenc(v));
  if (f == 0) atomicAdd(&cnt[bg], 1.0f);
}
__global__ void k_pool_fin(const float* cnt, const float* macc, const unsigned* pmax,
                           const float* sinfo, float* gvec, int G){
  long i = (long)blockIdx.x * blockDim.x + threadIdx.x;
  if (i >= (long)G * 259) return;
  int gi = (int)(i / 259), f = (int)(i % 259);
  float v;
  if (f < 128) {
    float c = cnt[gi]; if (c < 1.f) c = 1.f;
    v = macc[(size_t)gi * 128 + f] / c;
  } else if (f < 256) {
    float m = fdec(pmax[(size_t)gi * 128 + (f - 128)]);
    v = __builtin_isfinite(m) ? m : 0.f;
  } else {
    v = sinfo[(size_t)gi * 3 + (f - 256)];
  }
  gvec[(size_t)gi * 259 + f] = v;
}
__global__ void k_proj1(const float* gvec, const float* w1, const float* b1, float* hid, int G){
  long i = (long)blockIdx.x * blockDim.x + threadIdx.x;
  if (i >= (long)G * 128) return;
  int gi = (int)(i / 128), j = (int)(i % 128);
  float s = b1[j];
  const float* gv = gvec + (size_t)gi * 259;
  for (int k = 0; k < 259; ++k) s += gv[k] * w1[k * 128 + j];
  hid[(size_t)gi * 128 + j] = leakyf(s);
}
__global__ void k_proj2(const float* hid, const float* w2, const float* b2, float* out, int G){
  long i = (long)blockIdx.x * blockDim.x + threadIdx.x;
  if (i >= (long)G * 128) return;
  int gi = (int)(i / 128), j = (int)(i % 128);
  float s = b2[j];
  const float* hv = hid + (size_t)gi * 128;
  for (int k = 0; k < 128; ++k) s += hv[k] * w2[k * 128 + j];
  out[(size_t)gi * 128 + j] = s;
}

static inline unsigned nblk(long n, int t){ return (unsigned)((n + t - 1) / t); }

extern "C" void kernel_launch(void* const* d_in, const int* in_sizes, int n_in,
                              void* d_out, int out_size, void* d_ws, size_t ws_size,
                              hipStream_t stream) {
  const int*   eidx  = (const int*)  d_in[0];
  const int*   ridx  = (const int*)  d_in[1];
  const int*   didx  = (const int*)  d_in[2];
  const int*   eix   = (const int*)  d_in[3];
  const int*   batch = (const int*)  d_in[4];
  const float* avg   = (const float*)d_in[5];
  const float* sinfo = (const float*)d_in[6];
  const float* embE  = (const float*)d_in[7];
  const float* embR  = (const float*)d_in[8];
  const float* embD  = (const float*)d_in[9];
  const float* pw1   = (const float*)d_in[10];
  const float* pb1   = (const float*)d_in[11];
  const float* pw2   = (const float*)d_in[12];
  const float* pb2   = (const float*)d_in[13];
  const float* g0W   = (const float*)d_in[14];
  const float* g0as  = (const float*)d_in[15];
  const float* g0ad  = (const float*)d_in[16];
  const float* g0b   = (const float*)d_in[17];
  const float* g1W   = (const float*)d_in[18];
  const float* g1as  = (const float*)d_in[19];
  const float* g1ad  = (const float*)d_in[20];
  const float* g1b   = (const float*)d_in[21];
  const float* g2W   = (const float*)d_in[22];
  const float* g2as  = (const float*)d_in[23];
  const float* g2ad  = (const float*)d_in[24];
  const float* g2b   = (const float*)d_in[25];
  const float* ln0g  = (const float*)d_in[26];
  const float* ln0b  = (const float*)d_in[27];
  const float* ln1g  = (const float*)d_in[28];
  const float* ln1b  = (const float*)d_in[29];
  const float* ln2g  = (const float*)d_in[30];
  const float* ln2b  = (const float*)d_in[31];
  const float* pjw1  = (const float*)d_in[32];
  const float* pjb1  = (const float*)d_in[33];
  const float* pjw2  = (const float*)d_in[34];
  const float* pjb2  = (const float*)d_in[35];
  float* outp = (float*)d_out;

  const int N = in_sizes[0];
  const int E = in_sizes[3] / 2;
  const int G = in_sizes[6] / 3;
  const int Etot = E + N;
  const int* esrc = eix;
  const int* edst = eix + E;

  char* ws = (char*)d_ws;
  size_t off = 0;
  auto carve = [&](size_t bytes) -> char* {
    char* p = ws + off; off += (bytes + 255) & ~(size_t)255; return p;
  };
  unsigned short* Abf = (unsigned short*)carve((size_t)N * 512 * 2); // bf16 activations
  float*    Hf   = (float*)   carve((size_t)N * 512 * 4);            // GEMM out / h
  float*    Of   = (float*)   carve((size_t)N * 512 * 4);            // attn out (also initial x)
  float*    X0   = (float*)   carve((size_t)N * 512 * 4);            // post-LN activations
  unsigned short* Wt0 = (unsigned short*)carve((size_t)288 * 512 * 2);
  unsigned short* Wt1 = (unsigned short*)carve((size_t)512 * 512 * 2);
  unsigned short* Wt2 = (unsigned short*)carve((size_t)512 * 128 * 2);
  float*    asc  = (float*)   carve((size_t)N * 4 * 4);
  float*    dsc  = (float*)   carve((size_t)N * 4 * 4);
  unsigned* menc = (unsigned*)carve((size_t)N * 4 * 4);
  float*    ssum = (float*)   carve((size_t)N * 4 * 4);
  float*    cnt  = (float*)   carve((size_t)G * 4);
  float*    macc = (float*)   carve((size_t)G * 128 * 4);
  unsigned* pmax = (unsigned*)carve((size_t)G * 128 * 4);
  float*    gvec = (float*)   carve((size_t)G * 259 * 4);
  float*    hid  = (float*)   carve((size_t)G * 128 * 4);
  float*    X2   = (float*)Abf;  // alias: Abf dead after last GEMM

  const int T = 256;

  // weight transposes (bf16) once up front
  k_w_transpose_bf16<<<nblk((long)288 * 512, T), T, 0, stream>>>(g0W, Wt0, 288, 512);
  k_w_transpose_bf16<<<nblk((long)512 * 512, T), T, 0, stream>>>(g1W, Wt1, 512, 512);
  k_w_transpose_bf16<<<nblk((long)512 * 128, T), T, 0, stream>>>(g2W, Wt2, 512, 128);

  // node features -> Of[N,288], then bf16
  k_build_x<<<nblk((long)N * 288, T), T, 0, stream>>>(eidx, ridx, didx, avg, embE, embR,
                                                      embD, pw1, pb1, pw2, pb2, Of, N);
  k_f32_to_bf16<<<nblk((long)N * 288, T), T, 0, stream>>>(Of, Abf, (long)N * 288);

  // ---- layer 0: 288 -> 512 (4 heads x 128) ----
  {
    long waves = (long)((N + 31) / 32) * (512 / 32);
    k_wmma_gemm<<<nblk(waves * 32, T), T, 0, stream>>>(Abf, Wt0, Hf, N, 288, 512);
    k_scores<<<nblk((long)N * 4, T), T, 0, stream>>>(Hf, g0as, asc, N, 4, 128);
    k_scores<<<nblk((long)N * 4, T), T, 0, stream>>>(Hf, g0ad, dsc, N, 4, 128);
    k_fill_u32<<<nblk((long)N * 4, T), T, 0, stream>>>(menc, NEG_INF_ENC, (long)N * 4);
    k_fill_f32<<<nblk((long)N * 4, T), T, 0, stream>>>(ssum, 0.f, (long)N * 4);
    k_fill_f32<<<nblk((long)N * 512, T), T, 0, stream>>>(Of, 0.f, (long)N * 512);
    k_edge_max<<<nblk((long)Etot * 4, T), T, 0, stream>>>(esrc, edst, E, N, 4, asc, dsc, menc);
    k_edge_sum<<<nblk((long)Etot * 4, T), T, 0, stream>>>(esrc, edst, E, N, 4, asc, dsc, menc, ssum);
    k_edge_agg<<<nblk((long)Etot * 4, T), T, 0, stream>>>(esrc, edst, E, N, 4, 128, asc, dsc,
                                                          menc, ssum, Hf, Of);
    k_ln<<<nblk((long)N * 32, T), T, 0, stream>>>(Of, g0b, ln0g, ln0b, nullptr, X0, N, 512, 1, 0);
  }

  // ---- layer 1: 512 -> 512, residual ----
  {
    k_f32_to_bf16<<<nblk((long)N * 512, T), T, 0, stream>>>(X0, Abf, (long)N * 512);
    long waves = (long)((N + 31) / 32) * (512 / 32);
    k_wmma_gemm<<<nblk(waves * 32, T), T, 0, stream>>>(Abf, Wt1, Hf, N, 512, 512);
    k_scores<<<nblk((long)N * 4, T), T, 0, stream>>>(Hf, g1as, asc, N, 4, 128);
    k_scores<<<nblk((long)N * 4, T), T, 0, stream>>>(Hf, g1ad, dsc, N, 4, 128);
    k_fill_u32<<<nblk((long)N * 4, T), T, 0, stream>>>(menc, NEG_INF_ENC, (long)N * 4);
    k_fill_f32<<<nblk((long)N * 4, T), T, 0, stream>>>(ssum, 0.f, (long)N * 4);
    k_fill_f32<<<nblk((long)N * 512, T), T, 0, stream>>>(Of, 0.f, (long)N * 512);
    k_edge_max<<<nblk((long)Etot * 4, T), T, 0, stream>>>(esrc, edst, E, N, 4, asc, dsc, menc);
    k_edge_sum<<<nblk((long)Etot * 4, T), T, 0, stream>>>(esrc, edst, E, N, 4, asc, dsc, menc, ssum);
    k_edge_agg<<<nblk((long)Etot * 4, T), T, 0, stream>>>(esrc, edst, E, N, 4, 128, asc, dsc,
                                                          menc, ssum, Hf, Of);
    // x1 = leaky(ln(...)) + x0   (in-place over X0; elementwise-safe)
    k_ln<<<nblk((long)N * 32, T), T, 0, stream>>>(Of, g1b, ln1g, ln1b, X0, X0, N, 512, 1, 1);
  }

  // ---- layer 2: 512 -> 128, 1 head, no act/residual ----
  {
    k_f32_to_bf16<<<nblk((long)N * 512, T), T, 0, stream>>>(X0, Abf, (long)N * 512);
    long waves = (long)((N + 31) / 32) * (128 / 32);
    k_wmma_gemm<<<nblk(waves * 32, T), T, 0, stream>>>(Abf, Wt2, Hf, N, 512, 128);
    k_scores<<<nblk((long)N, T), T, 0, stream>>>(Hf, g2as, asc, N, 1, 128);
    k_scores<<<nblk((long)N, T), T, 0, stream>>>(Hf, g2ad, dsc, N, 1, 128);
    k_fill_u32<<<nblk((long)N, T), T, 0, stream>>>(menc, NEG_INF_ENC, (long)N);
    k_fill_f32<<<nblk((long)N, T), T, 0, stream>>>(ssum, 0.f, (long)N);
    k_fill_f32<<<nblk((long)N * 128, T), T, 0, stream>>>(Of, 0.f, (long)N * 128);
    k_edge_max<<<nblk((long)Etot, T), T, 0, stream>>>(esrc, edst, E, N, 1, asc, dsc, menc);
    k_edge_sum<<<nblk((long)Etot, T), T, 0, stream>>>(esrc, edst, E, N, 1, asc, dsc, menc, ssum);
    k_edge_agg<<<nblk((long)Etot, T), T, 0, stream>>>(esrc, edst, E, N, 1, 128, asc, dsc,
                                                      menc, ssum, Hf, Of);
    k_ln<<<nblk((long)N * 32, T), T, 0, stream>>>(Of, g2b, ln2g, ln2b, nullptr, X2, N, 128, 0, 0);
  }

  // ---- pooling + projection ----
  k_fill_f32<<<nblk((long)G, T), T, 0, stream>>>(cnt, 0.f, (long)G);
  k_fill_f32<<<nblk((long)G * 128, T), T, 0, stream>>>(macc, 0.f, (long)G * 128);
  k_fill_u32<<<nblk((long)G * 128, T), T, 0, stream>>>(pmax, NEG_INF_ENC, (long)G * 128);
  k_pool_acc<<<nblk((long)N * 128, T), T, 0, stream>>>(X2, batch, cnt, macc, pmax, N);
  k_pool_fin<<<nblk((long)G * 259, T), T, 0, stream>>>(cnt, macc, pmax, sinfo, gvec, G);
  k_proj1<<<nblk((long)G * 128, T), T, 0, stream>>>(gvec, pjw1, pjb1, hid, G);
  k_proj2<<<nblk((long)G * 128, T), T, 0, stream>>>(hid, pjw2, pjb2, outp, G);
}
